// DSAttention_37203006718710
// MI455X (gfx1250) — matrix operand order repaired and checked by reading, not compiled, not run on hardware
//
#include <hip/hip_runtime.h>
#include <math.h>

// ---------------- problem constants ----------------
#define BB 4
#define LL 2048
#define SS 2048
#define HH 8
#define EE 64
#define HE 512          // H*E
#define DM 512

typedef __attribute__((ext_vector_type(16))) _Float16 v16h;
typedef __attribute__((ext_vector_type(8)))  _Float16 v8h;
typedef __attribute__((ext_vector_type(8)))  float    v8f;

__device__ __forceinline__ v16h cat8(v8h a, v8h b) {
  return __builtin_shufflevector(a, b, 0,1,2,3,4,5,6,7,8,9,10,11,12,13,14,15);
}

__device__ __forceinline__ v8f wmma16(v16h a, v16h b, v8f c) {
  // D = A(16x32 f16) * B(32x16 f16) + C(16x16 f32)
  return __builtin_amdgcn_wmma_f32_16x16x32_f16(false, a, false, b, (short)0, c, false, false);
}

// ---------------------------------------------------------------------------
// Kernel 1: pack Q,K -> f16 [b][h][l][e]; V -> f16 transposed [b][h][e][s]
// ---------------------------------------------------------------------------
__global__ __launch_bounds__(256)
void pack_kernel(const float* __restrict__ Q, const float* __restrict__ K,
                 const float* __restrict__ V,
                 _Float16* __restrict__ Qh, _Float16* __restrict__ Kh,
                 _Float16* __restrict__ Vt) {
  const int lt = blockIdx.x;          // L/64 tiles
  const int h  = blockIdx.y;
  const int b  = blockIdx.z;
  const int tid = threadIdx.x;
  const int l0 = lt * 64;
  __shared__ _Float16 vtile[64][65];

#pragma unroll
  for (int i = 0; i < 16; ++i) {
    int lin = tid + i * 256;
    int lr = lin >> 6, le = lin & 63;
    size_t gin  = (((size_t)(b * SS + l0 + lr)) * HH + h) * EE + le;   // (B,S,H,E)
    size_t gout = (((size_t)(b * HH + h)) * LL + l0 + lr) * EE + le;   // (B,H,L,E)
    Qh[gout] = (_Float16)Q[gin];
    Kh[gout] = (_Float16)K[gin];
    vtile[lr][le] = (_Float16)V[gin];
  }
  __syncthreads();
#pragma unroll
  for (int i = 0; i < 16; ++i) {
    int lin = tid + i * 256;
    int er = lin >> 6, lc = lin & 63;
    Vt[(((size_t)(b * HH + h)) * EE + er) * SS + l0 + lc] = vtile[lc][er];
  }
}

// ---------------------------------------------------------------------------
// Kernel 2: mean/std over seq axis of values -> [b][h*E+e]
// ---------------------------------------------------------------------------
__global__ __launch_bounds__(256)
void stats_kernel(const float* __restrict__ V,
                  float* __restrict__ meanp, float* __restrict__ stdp) {
  int t = blockIdx.x * 256 + threadIdx.x;       // 0 .. B*HE-1
  int b = t >> 9, he = t & 511;
  const float* p = V + (size_t)b * SS * HE + he;
  float s = 0.f, ss = 0.f;
  for (int i = 0; i < SS; ++i) { float v = p[(size_t)i * HE]; s += v; ss += v * v; }
  float mu  = s * (1.f / (float)SS);
  float var = ss * (1.f / (float)SS) - mu * mu;      // biased (ddof=0)
  meanp[t] = mu;
  stdp[t]  = sqrtf(fmaxf(var, 0.f) + 1e-5f);
}

// ---------------------------------------------------------------------------
// Kernel 3: projector (circular conv over seq-channels + 4-layer MLP)
// x[b][c][j] with c in [0,2048) (permuted-seq channel), j in [0,512):
//   flat f = c*512+j ; h=f>>17 ; s=((c&255)<<3)|(j>>6) ; e=j&63
//   -> values offset (b*S+s)*512 + h*64 + e
// ---------------------------------------------------------------------------
__global__ __launch_bounds__(256)
void proj_kernel(const float* __restrict__ V,
                 const float* __restrict__ conv_w,                      // (2048,3)
                 const float* __restrict__ w1, const float* __restrict__ b1,
                 const float* __restrict__ w2, const float* __restrict__ b2,
                 const float* __restrict__ w3, const float* __restrict__ b3,
                 const float* __restrict__ w4,
                 const float* __restrict__ stats,                       // [B][512]
                 int outdim, float* __restrict__ outp, int apply_exp) {
  const int b = blockIdx.x;
  const int tid = threadIdx.x;
  __shared__ float cw[2048 * 3];
  __shared__ float xcat[1024];
  __shared__ float h1[512];
  __shared__ float h2[256];
  __shared__ float h3[128];

  for (int i = tid; i < 2048 * 3; i += 256) cw[i] = conv_w[i];
  __syncthreads();

  // circular conv: out width 512, kernel 3, channels 2048
  for (int w = tid; w < 512; w += 256) {
    int jm = (w + 511) & 511, j0 = w, jp = (w + 1) & 511;
    float acc = 0.f;
    for (int c = 0; c < 2048; ++c) {
      int hh = c >> 8;
      int sb = (c & 255) << 3;
      const float* base = V + ((size_t)b * SS) * HE + hh * EE;
      float xm = base[(size_t)(sb + (jm >> 6)) * HE + (jm & 63)];
      float x0 = base[(size_t)(sb + (j0 >> 6)) * HE + (j0 & 63)];
      float xp = base[(size_t)(sb + (jp >> 6)) * HE + (jp & 63)];
      const float* w3c = cw + c * 3;
      acc += xm * w3c[0] + x0 * w3c[1] + xp * w3c[2];
    }
    xcat[w] = acc;
  }
  for (int i = tid; i < 512; i += 256) xcat[512 + i] = stats[b * 512 + i];
  __syncthreads();

  for (int j = tid; j < 512; j += 256) {
    float acc = b1[j];
    for (int i = 0; i < 1024; ++i) acc += xcat[i] * w1[i * 512 + j];
    h1[j] = fmaxf(acc, 0.f);
  }
  __syncthreads();
  if (tid < 256) {
    float acc = b2[tid];
    for (int i = 0; i < 512; ++i) acc += h1[i] * w2[i * 256 + tid];
    h2[tid] = fmaxf(acc, 0.f);
  }
  __syncthreads();
  if (tid < 128) {
    float acc = b3[tid];
    for (int i = 0; i < 256; ++i) acc += h2[i] * w3[i * 128 + tid];
    h3[tid] = fmaxf(acc, 0.f);
  }
  __syncthreads();
  for (int j = tid; j < outdim; j += 256) {
    float acc = 0.f;
    for (int i = 0; i < 128; ++i) acc += h3[i] * w4[i * outdim + j];
    outp[b * outdim + j] = apply_exp ? __expf(acc) : acc;
  }
}

// ---------------------------------------------------------------------------
// One 32-key chunk of the flash-attention loop.
// MASKED=false: strictly-lower-triangular chunk (no causal compares).
// MASKED=true : diagonal chunk (per-element causal select).
// Row max:  4-step ds butterfly over the 16-lane half-groups.
// Row sum:  P x ones(32x16) on the WMMA pipe (replicated across lanes in
//           C layout), so numerator and denominator use the same f16 P.
// ---------------------------------------------------------------------------
template <bool MASKED>
__device__ __forceinline__ void attn_chunk(
    int s0, int q0, int n16, int hi, int kbase, int koff,
    float tsc, float sc, const float* __restrict__ dptr,
    const _Float16* __restrict__ kb_base, const _Float16* __restrict__ vt_base,
    const v16h (&qa)[4], v16h onesb,
    float (&m)[16], float (&l)[16], v8f (&o)[8],
    _Float16 (*pls)[32]) {
  // B fragments of K^T: lane = key col, halves = K(=e) koff..koff+15
  const _Float16* kr0 = kb_base + (size_t)(s0 + n16) * EE;
  const _Float16* kr1 = kb_base + (size_t)(s0 + 16 + n16) * EE;
  v16h kb0 = cat8(*(const v8h*)(kr0 + koff),      *(const v8h*)(kr0 + koff + 8));
  v16h kb1 = cat8(*(const v8h*)(kr0 + 32 + koff), *(const v8h*)(kr0 + 40 + koff));
  v16h kb2 = cat8(*(const v8h*)(kr1 + koff),      *(const v8h*)(kr1 + koff + 8));
  v16h kb3 = cat8(*(const v8h*)(kr1 + 32 + koff), *(const v8h*)(kr1 + 40 + koff));

  if (!MASKED) {  // speculative L2 prefetch of next chunk (always valid here)
    __builtin_prefetch(kb_base + (size_t)(s0 + 32 + n16) * EE, 0, 0);
    __builtin_prefetch(vt_base + (size_t)n16 * SS + s0 + 32, 0, 0);
  }

  // scores: c[2g+j] = Qgroup_g x Ktile_j  (K fragments reused by both groups)
  v8f c[4];
#pragma unroll
  for (int t = 0; t < 4; ++t) c[t] = (v8f){};
  c[0] = wmma16(qa[0], kb0, c[0]);  c[0] = wmma16(qa[1], kb1, c[0]);
  c[1] = wmma16(qa[0], kb2, c[1]);  c[1] = wmma16(qa[1], kb3, c[1]);
  c[2] = wmma16(qa[2], kb0, c[2]);  c[2] = wmma16(qa[3], kb1, c[2]);
  c[3] = wmma16(qa[2], kb2, c[3]);  c[3] = wmma16(qa[3], kb3, c[3]);

  float d0 = dptr[s0 + n16] * sc;        // scale folded into delta once/chunk
  float d1 = dptr[s0 + 16 + n16] * sc;

#pragma unroll
  for (int g = 0; g < 2; ++g) {
    float p0v[8], p1v[8], t8[8];
#pragma unroll
    for (int r = 0; r < 8; ++r) {
      float a0 = fmaf(c[2 * g][r], tsc, d0);          // single FMA per element
      float a1 = fmaf(c[2 * g + 1][r], tsc, d1);
      if (MASKED) {
        int ql = q0 + 16 * g + r + hi * 8;            // C layout: M = r + 8*hi
        if (s0 + n16 > ql)      a0 = -1e30f;
        if (s0 + 16 + n16 > ql) a1 = -1e30f;
      }
      p0v[r] = a0; p1v[r] = a1;
      t8[r] = fmaxf(a0, a1);
    }
#pragma unroll
    for (int r = 0; r < 8; ++r) {                     // row max over 16 lanes
      float v = t8[r];
#pragma unroll
      for (int off = 1; off < 16; off <<= 1) v = fmaxf(v, __shfl_xor(v, off, 32));
      t8[r] = v;
    }
#pragma unroll
    for (int r = 0; r < 8; ++r) {
      int ri = 8 * g + r;
      float mn   = fmaxf(m[ri], t8[r]);
      float corr = exp2f(m[ri] - mn);
      float p0 = exp2f(p0v[r] - mn);
      float p1 = exp2f(p1v[r] - mn);
      l[ri] *= corr;                                  // sum added via WMMA below
      m[ri] = mn;
      o[4 * g + 0][r] *= corr; o[4 * g + 1][r] *= corr;
      o[4 * g + 2][r] *= corr; o[4 * g + 3][r] *= corr;
      int mr = 16 * g + r + hi * 8;
      pls[mr][n16]      = (_Float16)p0;               // C-layout -> LDS [m][k]
      pls[mr][16 + n16] = (_Float16)p1;
    }
  }
  // LDS ops within a wave are in-order (DScnt): safe to read back now.
  v16h pa0 = cat8(*(const v8h*)&pls[n16][koff],      *(const v8h*)&pls[n16][koff + 8]);
  v16h pa1 = cat8(*(const v8h*)&pls[16 + n16][koff], *(const v8h*)&pls[16 + n16][koff + 8]);

  // softmax denominator on the matrix pipe: rowsum(P) = P x ones(32x16)
  {
    v8f z0 = (v8f){}, z1 = (v8f){};
    v8f ls0 = wmma16(pa0, onesb, z0);
    v8f ls1 = wmma16(pa1, onesb, z1);
#pragma unroll
    for (int r = 0; r < 8; ++r) { l[r] += ls0[r]; l[8 + r] += ls1[r]; }
  }

  // P(16x32) x V(32x16) for 4 e-tiles x 2 query groups; Vt rows contiguous in s.
#pragma unroll
  for (int t = 0; t < 4; ++t) {
    const _Float16* vr = vt_base + (size_t)(16 * t + n16) * SS + s0;
    v16h vb = cat8(*(const v8h*)(vr + koff), *(const v8h*)(vr + koff + 8));
    o[t]     = wmma16(pa0, vb, o[t]);
    o[4 + t] = wmma16(pa1, vb, o[4 + t]);
  }
}

// ---------------------------------------------------------------------------
// Kernel 4: causal flash attention with WMMA.
// One wave32 per 32-query tile per (b,h). Per 32-key chunk: 8 WMMA (scores,
// two 16-row query groups sharing the K fragments) + 2 WMMA (row sums)
// + 8 WMMA (P·V). Online softmax in exp2 domain; mask-free steady-state
// loop + one masked diagonal chunk.
// ---------------------------------------------------------------------------
__global__ __launch_bounds__(32)
void attn_kernel(const _Float16* __restrict__ Qh, const _Float16* __restrict__ Kh,
                 const _Float16* __restrict__ Vt, const float* __restrict__ taup,
                 const float* __restrict__ deltap, float* __restrict__ out) {
  const int qt = blockIdx.x, h = blockIdx.y, b = blockIdx.z;
  const int q0 = qt * 32;
  const int lane = threadIdx.x;
  const int n16  = lane & 15;
  const int hi   = lane >> 4;
  const int kbase = hi ? 8 : 0;      // unused; A fragments also use koff
  const int koff  = hi ? 16 : 0;     // B-fragment K offset

  const float sc  = 0.125f * 1.44269504088896340736f;  // (1/sqrt(64)) * log2(e)
  const float tsc = taup[b] * sc;                      // tau * scale, in SGPR
  const float* dptr = deltap + b * SS;

  // Q in A layout (16x32 f16) for both query groups
  v16h qa[4];
#pragma unroll
  for (int g = 0; g < 2; ++g) {
    const _Float16* qrow = Qh + (((size_t)(b * HH + h)) * LL + (q0 + 16 * g + n16)) * EE;
    qa[2 * g]     = cat8(*(const v8h*)(qrow + koff),      *(const v8h*)(qrow + koff + 8));
    qa[2 * g + 1] = cat8(*(const v8h*)(qrow + 32 + koff), *(const v8h*)(qrow + 40 + koff));
  }

  v16h onesb;                         // all-ones B fragment for row sums
#pragma unroll
  for (int i = 0; i < 16; ++i) onesb[i] = (_Float16)1.0f;

  float m[16], l[16];
  v8f o[8];
#pragma unroll
  for (int r = 0; r < 16; ++r) { m[r] = -1e30f; l[r] = 0.f; }
#pragma unroll
  for (int t = 0; t < 8; ++t) o[t] = (v8f){};

  __shared__ __align__(16) _Float16 pls[32][32];

  const _Float16* kb_base = Kh + ((size_t)(b * HH + h)) * LL * EE;
  const _Float16* vt_base = Vt + ((size_t)(b * HH + h)) * EE * SS;

  // steady state: all keys strictly below the diagonal tile -> no masking
  for (int s0 = 0; s0 < q0; s0 += 32)
    attn_chunk<false>(s0, q0, n16, hi, kbase, koff, tsc, sc, dptr,
                      kb_base, vt_base, qa, onesb, m, l, o, pls);
  // diagonal 32x32 chunk with causal select
  attn_chunk<true>(q0, q0, n16, hi, kbase, koff, tsc, sc, dptr,
                   kb_base, vt_base, qa, onesb, m, l, o, pls);

  // epilogue: normalize and store (B,L,H,E) f32
#pragma unroll
  for (int g = 0; g < 2; ++g) {
#pragma unroll
    for (int r = 0; r < 8; ++r) {
      float inv = 1.0f / l[8 * g + r];
      int ql = q0 + 16 * g + r + hi * 8;
      float* orow = out + (((size_t)(b * LL) + ql) * HH + h) * EE;
      orow[n16]      = o[4 * g + 0][r] * inv;
      orow[16 + n16] = o[4 * g + 1][r] * inv;
      orow[32 + n16] = o[4 * g + 2][r] * inv;
      orow[48 + n16] = o[4 * g + 3][r] * inv;
    }
  }
}

// ---------------------------------------------------------------------------
extern "C" void kernel_launch(void* const* d_in, const int* in_sizes, int n_in,
                              void* d_out, int out_size, void* d_ws, size_t ws_size,
                              hipStream_t stream) {
  const float* queries     = (const float*)d_in[0];
  const float* keys        = (const float*)d_in[1];
  const float* values      = (const float*)d_in[2];
  const float* tau_conv_w  = (const float*)d_in[3];
  const float* tau_w1      = (const float*)d_in[4];
  const float* tau_b1      = (const float*)d_in[5];
  const float* tau_w2      = (const float*)d_in[6];
  const float* tau_b2      = (const float*)d_in[7];
  const float* tau_w3      = (const float*)d_in[8];
  const float* tau_b3      = (const float*)d_in[9];
  const float* tau_w4      = (const float*)d_in[10];
  const float* del_conv_w  = (const float*)d_in[11];
  const float* del_w1      = (const float*)d_in[12];
  const float* del_b1      = (const float*)d_in[13];
  const float* del_w2      = (const float*)d_in[14];
  const float* del_b2      = (const float*)d_in[15];
  const float* del_w3      = (const float*)d_in[16];
  const float* del_b3      = (const float*)d_in[17];
  const float* del_w4      = (const float*)d_in[18];

  // workspace layout (~25.2 MB)
  char* ws = (char*)d_ws;
  const size_t TEN = (size_t)BB * HH * LL * EE * sizeof(_Float16);   // 8 MiB each
  _Float16* Qh   = (_Float16*)(ws);
  _Float16* Kh   = (_Float16*)(ws + TEN);
  _Float16* Vt   = (_Float16*)(ws + 2 * TEN);
  float*    mnp  = (float*)(ws + 3 * TEN);
  float*    stp  = (float*)(ws + 3 * TEN + 8192);
  float*    taup = (float*)(ws + 3 * TEN + 16384);
  float*    delp = (float*)(ws + 3 * TEN + 16384 + 256);

  pack_kernel<<<dim3(LL / 64, HH, BB), 256, 0, stream>>>(queries, keys, values, Qh, Kh, Vt);
  stats_kernel<<<(BB * HE) / 256, 256, 0, stream>>>(values, mnp, stp);
  proj_kernel<<<BB, 256, 0, stream>>>(values, tau_conv_w, tau_w1, tau_b1, tau_w2, tau_b2,
                                      tau_w3, tau_b3, tau_w4, stp, 1, taup, 1);
  proj_kernel<<<BB, 256, 0, stream>>>(values, del_conv_w, del_w1, del_b1, del_w2, del_b2,
                                      del_w3, del_b3, del_w4, mnp, SS, delp, 0);
  attn_kernel<<<dim3(LL / 32, HH, BB), 32, 0, stream>>>(Qh, Kh, Vt, taup, delp, (float*)d_out);
}
